// MultiHeadAttention_20624432955931
// MI455X (gfx1250) — matrix operand order repaired and checked by reading, not compiled, not run on hardware
//
#include <hip/hip_runtime.h>
#include <hip/hip_bf16.h>

// ---------------------------------------------------------------------------
// MHA forward for MI455X (gfx1250, wave32, WMMA).
// bf16 operands + f32 WMMA accumulation (V_WMMA_F32_16X16X32_BF16):
// keeps compute near the 512MB attn-write memory floor (~22us @ 23.3 TB/s).
// This revision: amdgpu_waves_per_eu register-budget hints so the allocator
// keeps BOTH ping-pong operand buffer sets live (round-3 asm showed it
// recycling one B-buffer register set, re-serializing load->wait->WMMA).
// Workspace use: ~48 MB.
// ---------------------------------------------------------------------------

typedef __bf16 bf16;
typedef __attribute__((ext_vector_type(16))) __bf16 v16bf;
typedef __attribute__((ext_vector_type(8)))  __bf16 v8bf;
typedef __attribute__((ext_vector_type(8)))  float  v8f;

#define DEV static __device__ __forceinline__

DEV v16bf join8(v8bf a, v8bf b) {
  return __builtin_shufflevector(a, b, 0,1,2,3,4,5,6,7,8,9,10,11,12,13,14,15);
}
DEV v8f wmma_bf16(v16bf a, v16bf b, v8f c) {
  // (neg_a, A, neg_b, B, c_mod, C, reuse_a, reuse_b)
  return __builtin_amdgcn_wmma_f32_16x16x32_bf16(false, a, false, b, (short)0, c,
                                                 false, false);
}

constexpr int NB    = 2;     // batch
constexpr int NH    = 16;    // heads
constexpr int SEQ   = 2048;  // sequence length
constexpr int HD    = 64;    // per-head dim
constexpr int DMODEL= 1024;  // model dim

// ---------------------------------------------------------------------------
// Prep kernels
// ---------------------------------------------------------------------------
__global__ void cvt_bf16(const float* __restrict__ s, bf16* __restrict__ d, int n) {
  int i = blockIdx.x * blockDim.x + threadIdx.x;
  if (i < n) d[i] = (bf16)s[i];
}

// WT[n][k] = W[k][n]  (bf16), W is DMODEL x DMODEL f32, LDS-tiled transpose
__global__ void transpose_bf16(const float* __restrict__ W, bf16* __restrict__ WT) {
  __shared__ float tile[32][33];
  int tx = threadIdx.x, ty = threadIdx.y;
  int c0 = blockIdx.x * 32, r0 = blockIdx.y * 32;
  for (int j = 0; j < 32; j += 8)
    tile[ty + j][tx] = W[(size_t)(r0 + ty + j) * DMODEL + c0 + tx];
  __syncthreads();
  for (int j = 0; j < 32; j += 8)
    WT[(size_t)(c0 + ty + j) * DMODEL + r0 + tx] = (bf16)tile[tx][ty + j];
}

// WoT[n][h*64+d] = Wo[d*16+h][n] : fold the reference's (B,L,DQ,H) concat
// permutation into Wo so out_heads can be stored in standard [l][h*64+d].
__global__ void wo_prep(const float* __restrict__ Wo, bf16* __restrict__ WoT) {
  int i = blockIdx.x * blockDim.x + threadIdx.x;   // [0, DMODEL*DMODEL)
  int n  = i >> 10;
  int kk = i & 1023;
  int hh = kk >> 6;
  int d  = kk & 63;
  WoT[i] = (bf16)Wo[((size_t)(d * NH + hh)) * DMODEL + n];
}

// ---------------------------------------------------------------------------
// Fused QKV projection GEMM (ping-pong buffered): C = xbf @ W{q,k,v}T^T + bias
//   grid = (M/16, N/512, 3), block = 256 (8 waves), wave -> 16x64 tile
//   z==0 -> Qbf[b][h][l][dq], z==1 -> Kbf same, z==2 -> Vt[b][h][dq][l]
// ---------------------------------------------------------------------------
__global__ void __launch_bounds__(256)
__attribute__((amdgpu_waves_per_eu(4, 8)))
qkv_gemm(
    const bf16* __restrict__ xbf,
    const bf16* __restrict__ WqT, const bf16* __restrict__ WkT,
    const bf16* __restrict__ WvT,
    const float* __restrict__ bq, const float* __restrict__ bk,
    const float* __restrict__ bv,
    bf16* __restrict__ Qbf, bf16* __restrict__ Kbf, bf16* __restrict__ Vt)
{
  int wave = threadIdx.x >> 5, lane = threadIdx.x & 31;
  int laneN = lane & 15, hi = lane >> 4;
  int m0 = blockIdx.x * 16;
  int n0 = blockIdx.y * 512 + wave * 64;
  int z  = blockIdx.z;
  const bf16*  WT   = (z == 0) ? WqT : (z == 1) ? WkT : WvT;
  const float* bias = (z == 0) ? bq  : (z == 1) ? bk  : bv;

  v8f acc[4];
  #pragma unroll
  for (int t = 0; t < 4; t++) for (int v = 0; v < 8; v++) acc[t][v] = 0.f;

  const bf16* arowp = xbf + (size_t)(m0 + laneN) * DMODEL + hi * 16;
  auto ldA = [&](int k0) {
    return join8(*(const v8bf*)(arowp + k0), *(const v8bf*)(arowp + k0 + 8));
  };
  auto ldB = [&](int t, int k0) {
    const bf16* brp = WT + (size_t)(n0 + t * 16 + laneN) * DMODEL + k0 + hi * 16;
    return join8(*(const v8bf*)brp, *(const v8bf*)(brp + 8));
  };

  // ping-pong buffer sets; no cross-iteration register copies
  v16bf aP = ldA(0);
  v16bf bP0 = ldB(0, 0), bP1 = ldB(1, 0), bP2 = ldB(2, 0), bP3 = ldB(3, 0);

  for (int k0 = 0; k0 < DMODEL; k0 += 64) {
    int k1 = k0 + 32;                              // always < DMODEL
    v16bf aQ = ldA(k1);
    v16bf bQ0 = ldB(0, k1), bQ1 = ldB(1, k1), bQ2 = ldB(2, k1), bQ3 = ldB(3, k1);
    acc[0] = wmma_bf16(aP, bP0, acc[0]);
    acc[1] = wmma_bf16(aP, bP1, acc[1]);
    acc[2] = wmma_bf16(aP, bP2, acc[2]);
    acc[3] = wmma_bf16(aP, bP3, acc[3]);
    int k2 = (k0 + 64 < DMODEL) ? k0 + 64 : 0;     // branchless tail reload
    aP = ldA(k2);
    bP0 = ldB(0, k2); bP1 = ldB(1, k2); bP2 = ldB(2, k2); bP3 = ldB(3, k2);
    acc[0] = wmma_bf16(aQ, bQ0, acc[0]);
    acc[1] = wmma_bf16(aQ, bQ1, acc[1]);
    acc[2] = wmma_bf16(aQ, bQ2, acc[2]);
    acc[3] = wmma_bf16(aQ, bQ3, acc[3]);
  }

  #pragma unroll
  for (int t = 0; t < 4; t++) {
    int c  = n0 + t * 16 + laneN;
    int hh = c >> 6, dq = c & 63;
    #pragma unroll
    for (int v = 0; v < 8; v++) {
      int r = m0 + v + hi * 8;
      int bidx = r >> 11, l = r & 2047;
      bf16 o = (bf16)(acc[t][v] + bias[c]);
      if (z == 2)      Vt [(((size_t)(bidx * NH + hh)) * HD  + dq) * SEQ + l] = o;
      else if (z == 1) Kbf[(((size_t)(bidx * NH + hh)) * SEQ + l ) * HD  + dq] = o;
      else             Qbf[(((size_t)(bidx * NH + hh)) * SEQ + l ) * HD  + dq] = o;
    }
  }
}

// ---------------------------------------------------------------------------
// Fused attention: one workgroup (8 waves) per (b, h, 16-row query tile).
// Keys split 256/wave; full 16x2048 score block lives in VGPRs (128/wave).
// Score & PV loops: fully unrolled chunk-PAIRS with alternating (ping-pong)
// operand buffer sets and wave-uniform activity guards.
// ---------------------------------------------------------------------------
__global__ void __launch_bounds__(256)
__attribute__((amdgpu_waves_per_eu(2, 8)))
attn_fused(
    const bf16* __restrict__ Qbf, const bf16* __restrict__ Kbf,
    const bf16* __restrict__ Vt,
    bf16* __restrict__ Obf, float* __restrict__ attn)
{
  __shared__ float red[2][8][16];       // [max|sum][wave][row]
  __shared__ float obuf[8][64][17];     // padded: bank-conflict-free reduce

  int wave = threadIdx.x >> 5, lane = threadIdx.x & 31;
  int laneN = lane & 15, hi = lane >> 4;
  int rbase = hi * 8;                   // C-layout: VGPR v -> row v + rbase

  int tile = blockIdx.x, h = blockIdx.y, b = blockIdx.z;
  int m0 = tile * 16;
  int qlimit = m0 + 15;                 // largest unmasked key for this tile
  int ws = wave * 256;                  // this wave's key-slice start

  const bf16* Qh = Qbf + (((size_t)(b * NH + h)) * SEQ + m0) * HD;
  const bf16* Kh = Kbf + ((size_t)(b * NH + h)) * SEQ * HD;
  const bf16* Vh = Vt  + ((size_t)(b * NH + h)) * HD * SEQ;
  float*      Ah = attn + (((size_t)(b * NH + h)) * SEQ + m0) * SEQ;

  // number of active 16-key / 32-key chunks for this wave (wave-uniform)
  int nact  = (ws <= qlimit) ? (((qlimit - ws) >> 4) + 1) : 0;
  if (nact > 16) nact = 16;
  int nact2 = (ws <= qlimit) ? (((qlimit - ws) >> 5) + 1) : 0;
  if (nact2 > 8) nact2 = 8;

  // Warm L0/L2 for this wave's K and V^T slices (global_prefetch_b8).
  if (nact > 0) {
    #pragma unroll
    for (int i = 0; i < 8; i++)
      __builtin_prefetch(Kh + (size_t)(ws + i * 32 + lane) * HD, 0, 3);
    #pragma unroll
    for (int i = 0; i < 8; i++) {
      int li = i * 32 + lane;           // 256 cache lines: 64 rows x 4 segs
      __builtin_prefetch(Vh + (size_t)(li >> 2) * SEQ + ws + (li & 3) * 64, 0, 3);
    }
  }

  // A-operand: Q rows m0..m0+15, K-dim = 64 (two K=32 chunks)
  const bf16* qrow = Qh + (size_t)laneN * HD + hi * 16;
  v16bf a0 = join8(*(const v8bf*)(qrow),      *(const v8bf*)(qrow + 8));
  v16bf a1 = join8(*(const v8bf*)(qrow + 32), *(const v8bf*)(qrow + 40));

  auto ldK = [&](int j0, v16bf& o0, v16bf& o1) {
    const bf16* krow = Kh + (size_t)(j0 + laneN) * HD + hi * 16;
    o0 = join8(*(const v8bf*)(krow),      *(const v8bf*)(krow + 8));
    o1 = join8(*(const v8bf*)(krow + 32), *(const v8bf*)(krow + 40));
  };

  v8f zero8; for (int v = 0; v < 8; v++) zero8[v] = 0.f;

  v8f   sc[16];
  float vmax[8];
  #pragma unroll
  for (int v = 0; v < 8; v++) vmax[v] = -__builtin_inff();

  // scale + causal mask + running row-max for one chunk (compile-time c)
  auto finish = [&](v8f s, int c) {
    int key = ws + c * 16 + laneN;
    #pragma unroll
    for (int v = 0; v < 8; v++) {
      int qr = m0 + v + rbase;
      float val = s[v] * 0.125f;        // 1/sqrt(64), applied before masking
      if (key > qr) val = -__builtin_inff();
      s[v] = val;
      vmax[v] = fmaxf(vmax[v], val);
    }
    return s;
  };

  v16bf kA0, kA1, kB0, kB1;             // ping-pong K-operand buffers
  if (nact > 0) ldK(ws, kA0, kA1);

  #pragma unroll
  for (int cp = 0; cp < 8; cp++) {
    int c0 = 2 * cp, c1 = 2 * cp + 1;
    if (c0 < nact) {                    // wave-uniform; EXEC stays all-ones
      int jn = (c1 < nact) ? ws + c1 * 16 : ws;
      ldK(jn, kB0, kB1);                // buffer B fills during A's WMMAs
      v8f s = wmma_bf16(a0, kA0, zero8);
      s = wmma_bf16(a1, kA1, s);
      sc[c0] = finish(s, c0);
    } else {
      v8f s; for (int v = 0; v < 8; v++) s[v] = -__builtin_inff();
      sc[c0] = s;
    }
    if (c1 < nact) {
      int jn = (c1 + 1 < nact) ? ws + (c1 + 1) * 16 : ws;
      ldK(jn, kA0, kA1);                // buffer A refills during B's WMMAs
      v8f s = wmma_bf16(a0, kB0, zero8);
      s = wmma_bf16(a1, kB1, s);
      sc[c1] = finish(s, c1);
    } else {
      v8f s; for (int v = 0; v < 8; v++) s[v] = -__builtin_inff();
      sc[c1] = s;
    }
  }

  // row max: reduce across the 16 lanes of each half, then across waves (LDS)
  #pragma unroll
  for (int v = 0; v < 8; v++)
    for (int off = 1; off < 16; off <<= 1)
      vmax[v] = fmaxf(vmax[v], __shfl_xor(vmax[v], off, 32));
  if (laneN == 0)
    for (int v = 0; v < 8; v++) red[0][wave][v + rbase] = vmax[v];
  __syncthreads();
  float gmax[8];
  #pragma unroll
  for (int v = 0; v < 8; v++) {
    float m = red[0][0][v + rbase];
    for (int w = 1; w < 8; w++) m = fmaxf(m, red[0][w][v + rbase]);
    gmax[v] = m;
  }

  // exp + row sum
  float vsum[8]; for (int v = 0; v < 8; v++) vsum[v] = 0.f;
  #pragma unroll
  for (int c = 0; c < 16; c++) {
    #pragma unroll
    for (int v = 0; v < 8; v++) {
      float e = __expf(sc[c][v] - gmax[v]);   // exp(-inf)=0 in masked region
      sc[c][v] = e;
      vsum[v] += e;
    }
  }
  #pragma unroll
  for (int v = 0; v < 8; v++)
    for (int off = 1; off < 16; off <<= 1)
      vsum[v] += __shfl_xor(vsum[v], off, 32);
  if (laneN == 0)
    for (int v = 0; v < 8; v++) red[1][wave][v + rbase] = vsum[v];
  __syncthreads();
  float rcp[8];
  #pragma unroll
  for (int v = 0; v < 8; v++) {
    float sg = 0.f;
    for (int w = 0; w < 8; w++) sg += red[1][w][v + rbase];
    rcp[v] = 1.0f / sg;                 // >= 1, never zero (max term = 1)
  }

  // normalize and write attn (f32, includes exact zeros above the diagonal)
  #pragma unroll
  for (int c = 0; c < 16; c++) {
    size_t col = (size_t)(ws + c * 16 + laneN);
    #pragma unroll
    for (int v = 0; v < 8; v++) {
      int r = v + rbase;
      Ah[(size_t)r * SEQ + col] = sc[c][v] * rcp[v];
    }
  }

  // PV: D2[d][m] = sum_j V[j][d] * p[m][j].  A = V^T rows (from Vt, global),
  // B = p rows re-read from just-written attn (same-wave store->load ordered
  // by HW; L0-hot), f32->bf16 in VALU.  Ping-pong buffered chunk pairs.
  v8f acc[4];
  #pragma unroll
  for (int t = 0; t < 4; t++) for (int v = 0; v < 8; v++) acc[t][v] = 0.f;

  auto ldPV = [&](int jb, float4* f, v16bf* aa) {
    const float* pr = Ah + (size_t)laneN * SEQ + jb + hi * 16;
    f[0] = *(const float4*)(pr);
    f[1] = *(const float4*)(pr + 4);
    f[2] = *(const float4*)(pr + 8);
    f[3] = *(const float4*)(pr + 12);
    #pragma unroll
    for (int t = 0; t < 4; t++) {
      const bf16* vrow = Vh + (size_t)(t * 16 + laneN) * SEQ + jb + hi * 16;
      aa[t] = join8(*(const v8bf*)(vrow), *(const v8bf*)(vrow + 8));
    }
  };
  auto cvtP = [&](const float4* f) {
    v16bf bb;
    bb[0]=(bf16)f[0].x;  bb[1]=(bf16)f[0].y;  bb[2]=(bf16)f[0].z;  bb[3]=(bf16)f[0].w;
    bb[4]=(bf16)f[1].x;  bb[5]=(bf16)f[1].y;  bb[6]=(bf16)f[1].z;  bb[7]=(bf16)f[1].w;
    bb[8]=(bf16)f[2].x;  bb[9]=(bf16)f[2].y;  bb[10]=(bf16)f[2].z; bb[11]=(bf16)f[2].w;
    bb[12]=(bf16)f[3].x; bb[13]=(bf16)f[3].y; bb[14]=(bf16)f[3].z; bb[15]=(bf16)f[3].w;
    return bb;
  };

  float4 fA[4]; v16bf aA[4];
  float4 fB[4]; v16bf aB[4];
  if (nact2 > 0) ldPV(ws, fA, aA);

  #pragma unroll
  for (int kp = 0; kp < 4; kp++) {
    int k0 = 2 * kp, k1 = 2 * kp + 1;
    if (k0 < nact2) {                   // wave-uniform
      int jn = (k1 < nact2) ? ws + k1 * 32 : ws;
      ldPV(jn, fB, aB);
      v16bf bb = cvtP(fA);
      #pragma unroll
      for (int t = 0; t < 4; t++) acc[t] = wmma_bf16(aA[t], bb, acc[t]);
    }
    if (k1 < nact2) {
      int jn = (k1 + 1 < nact2) ? ws + (k1 + 1) * 32 : ws;
      ldPV(jn, fA, aA);
      v16bf bb = cvtP(fB);
      #pragma unroll
      for (int t = 0; t < 4; t++) acc[t] = wmma_bf16(aB[t], bb, acc[t]);
    }
  }

  // cross-wave reduction of D2 partials, write out_heads (standard layout;
  // the reference's permutation was folded into WoT).
  #pragma unroll
  for (int t = 0; t < 4; t++)
    for (int v = 0; v < 8; v++)
      obuf[wave][t * 16 + v + rbase][laneN] = acc[t][v];
  __syncthreads();

  int tt = threadIdx.x;
  int mq = tt >> 4;
  int dbase = (tt & 15) * 4;
  #pragma unroll
  for (int i = 0; i < 4; i++) {
    int d = dbase + i;
    float sum = 0.f;
    for (int w = 0; w < 8; w++) sum += obuf[w][d][mq];
    Obf[((size_t)(b * SEQ + m0 + mq)) * DMODEL + h * HD + d] = (bf16)sum;
  }
}

// ---------------------------------------------------------------------------
// Output projection (ping-pong buffered): out = Obf @ WoT^T + bo  (f32)
// ---------------------------------------------------------------------------
__global__ void __launch_bounds__(256)
__attribute__((amdgpu_waves_per_eu(4, 8)))
out_gemm(
    const bf16* __restrict__ Obf, const bf16* __restrict__ WoT,
    const float* __restrict__ bo, float* __restrict__ out)
{
  int wave = threadIdx.x >> 5, lane = threadIdx.x & 31;
  int laneN = lane & 15, hi = lane >> 4;
  int m0 = blockIdx.x * 16;
  int n0 = blockIdx.y * 512 + wave * 64;

  v8f acc[4];
  #pragma unroll
  for (int t = 0; t < 4; t++) for (int v = 0; v < 8; v++) acc[t][v] = 0.f;

  const bf16* arowp = Obf + (size_t)(m0 + laneN) * DMODEL + hi * 16;
  auto ldA = [&](int k0) {
    return join8(*(const v8bf*)(arowp + k0), *(const v8bf*)(arowp + k0 + 8));
  };
  auto ldB = [&](int t, int k0) {
    const bf16* brp = WoT + (size_t)(n0 + t * 16 + laneN) * DMODEL + k0 + hi * 16;
    return join8(*(const v8bf*)brp, *(const v8bf*)(brp + 8));
  };

  v16bf aP = ldA(0);
  v16bf bP0 = ldB(0, 0), bP1 = ldB(1, 0), bP2 = ldB(2, 0), bP3 = ldB(3, 0);

  for (int k0 = 0; k0 < DMODEL; k0 += 64) {
    int k1 = k0 + 32;
    v16bf aQ = ldA(k1);
    v16bf bQ0 = ldB(0, k1), bQ1 = ldB(1, k1), bQ2 = ldB(2, k1), bQ3 = ldB(3, k1);
    acc[0] = wmma_bf16(aP, bP0, acc[0]);
    acc[1] = wmma_bf16(aP, bP1, acc[1]);
    acc[2] = wmma_bf16(aP, bP2, acc[2]);
    acc[3] = wmma_bf16(aP, bP3, acc[3]);
    int k2 = (k0 + 64 < DMODEL) ? k0 + 64 : 0;
    aP = ldA(k2);
    bP0 = ldB(0, k2); bP1 = ldB(1, k2); bP2 = ldB(2, k2); bP3 = ldB(3, k2);
    acc[0] = wmma_bf16(aQ, bQ0, acc[0]);
    acc[1] = wmma_bf16(aQ, bQ1, acc[1]);
    acc[2] = wmma_bf16(aQ, bQ2, acc[2]);
    acc[3] = wmma_bf16(aQ, bQ3, acc[3]);
  }

  #pragma unroll
  for (int t = 0; t < 4; t++) {
    int c = n0 + t * 16 + laneN;
    #pragma unroll
    for (int v = 0; v < 8; v++) {
      int r = m0 + v + hi * 8;
      out[(size_t)r * DMODEL + c] = acc[t][v] + bo[c];
    }
  }
}

// ---------------------------------------------------------------------------
// Host launch
// ---------------------------------------------------------------------------
extern "C" void kernel_launch(void* const* d_in, const int* in_sizes, int n_in,
                              void* d_out, int out_size, void* d_ws, size_t ws_size,
                              hipStream_t stream) {
  (void)in_sizes; (void)n_in; (void)out_size; (void)ws_size;
  const float* x  = (const float*)d_in[0];
  const float* Wq = (const float*)d_in[1];
  const float* bq = (const float*)d_in[2];
  const float* Wk = (const float*)d_in[3];
  const float* bk = (const float*)d_in[4];
  const float* Wv = (const float*)d_in[5];
  const float* bv = (const float*)d_in[6];
  const float* Wo = (const float*)d_in[7];
  const float* bo = (const float*)d_in[8];

  float* out  = (float*)d_out;                         // [B, L, D]
  float* attn = out + (size_t)NB * SEQ * DMODEL;       // [B, H, L, L]

  char* p = (char*)d_ws;
  auto take = [&](size_t bytes) {
    char* r = p; p += (bytes + 255) & ~(size_t)255; return r;
  };
  bf16* xbf = (bf16*)take((size_t)NB * SEQ * DMODEL * 2);
  bf16* WqT = (bf16*)take((size_t)DMODEL * DMODEL * 2);
  bf16* WkT = (bf16*)take((size_t)DMODEL * DMODEL * 2);
  bf16* WvT = (bf16*)take((size_t)DMODEL * DMODEL * 2);
  bf16* WoT = (bf16*)take((size_t)DMODEL * DMODEL * 2);
  bf16* Qbf = (bf16*)take((size_t)NB * NH * SEQ * HD * 2);
  bf16* Kbf = (bf16*)take((size_t)NB * NH * SEQ * HD * 2);
  bf16* Vt  = (bf16*)take((size_t)NB * NH * SEQ * HD * 2);
  bf16* Obf = (bf16*)take((size_t)NB * SEQ * DMODEL * 2);

  int nx = NB * SEQ * DMODEL;
  cvt_bf16<<<nx / 256, 256, 0, stream>>>(x, xbf, nx);

  dim3 tb(32, 8);
  dim3 tg(DMODEL / 32, DMODEL / 32);
  transpose_bf16<<<tg, tb, 0, stream>>>(Wq, WqT);
  transpose_bf16<<<tg, tb, 0, stream>>>(Wk, WkT);
  transpose_bf16<<<tg, tb, 0, stream>>>(Wv, WvT);
  wo_prep<<<(DMODEL * DMODEL) / 256, 256, 0, stream>>>(Wo, WoT);

  qkv_gemm<<<dim3(NB * SEQ / 16, DMODEL / 512, 3), 256, 0, stream>>>(
      xbf, WqT, WkT, WvT, bq, bk, bv, Qbf, Kbf, Vt);

  attn_fused<<<dim3(SEQ / 16, NH, NB), 256, 0, stream>>>(
      Qbf, Kbf, Vt, Obf, attn);

  out_gemm<<<dim3(NB * SEQ / 16, DMODEL / 512), 256, 0, stream>>>(
      Obf, WoT, bo, out);
}
